// DeepSeekMoE_22239340658921
// MI455X (gfx1250) — matrix operand order repaired and checked by reading, not compiled, not run on hardware
//
#include <hip/hip_runtime.h>
#include <hip/hip_bf16.h>

// ---------------------------------------------------------------------------
// DeepSeek-style MoE (shared expert + 8 routed, top-2) for gfx1250 (MI455X).
// bf16 WMMA (v_wmma_f32_16x16x32_bf16); weights pre-transposed so B-fragments
// are contiguous 32B loads; bf16 weight set (151 MB) is L2-resident (192 MB).
// This revision: explicit A-fragment hoisting + 1-deep software pipelining of
// B-fragment global loads in both GEMM phases to overlap L2 latency with WMMA.
// ---------------------------------------------------------------------------

#define H_DIM   1024
#define I_DIM   4096
#define NEXP    8
#define T_TOK   16384        // 4 * 4096 tokens
#define M_TILE  32           // tokens per workgroup
#define IC      64           // intermediate-dim chunk
#define XLD     (H_DIM + 8)  // padded LDS stride (bank-conflict-free A loads)
#define ULD     (IC + 8)

typedef __bf16 bf16_t;
typedef __attribute__((ext_vector_type(16))) __bf16 v16bf;
typedef __attribute__((ext_vector_type(8)))  __bf16 v8bf;
typedef __attribute__((ext_vector_type(4)))  __bf16 v4bf;
typedef __attribute__((ext_vector_type(8)))  float  v8f;

union ABfrag { v16bf v; v8bf h[2]; };

// ---------------------------------------------------------------------------
// fp32 -> bf16 elementwise convert (vectorized x4)
// ---------------------------------------------------------------------------
__global__ void cvt_bf16_kernel(const float* __restrict__ src,
                                bf16_t* __restrict__ dst, size_t n4) {
    size_t i = (size_t)blockIdx.x * blockDim.x + threadIdx.x;
    if (i >= n4) return;
    float4 f = reinterpret_cast<const float4*>(src)[i];
    v4bf o;
    o[0] = (__bf16)f.x; o[1] = (__bf16)f.y;
    o[2] = (__bf16)f.z; o[3] = (__bf16)f.w;
    reinterpret_cast<v4bf*>(dst)[i] = o;
}

// ---------------------------------------------------------------------------
// fp32 [R][C] -> bf16 [C][R] transpose-convert, 32x32 LDS tiles, batched in z
// ---------------------------------------------------------------------------
__global__ void transpose_cvt_kernel(const float* __restrict__ src,
                                     bf16_t* __restrict__ dst,
                                     int R, int C) {
    __shared__ float tile[32][33];
    size_t mat = (size_t)blockIdx.z * (size_t)R * (size_t)C;
    src += mat; dst += mat;
    int c0 = blockIdx.x * 32, r0 = blockIdx.y * 32;
    #pragma unroll
    for (int i = threadIdx.y; i < 32; i += 8)
        tile[i][threadIdx.x] = src[(size_t)(r0 + i) * C + (c0 + threadIdx.x)];
    __syncthreads();
    #pragma unroll
    for (int i = threadIdx.y; i < 32; i += 8)
        dst[(size_t)(c0 + i) * R + (r0 + threadIdx.x)] =
            (bf16_t)tile[threadIdx.x][i];
}

// ---------------------------------------------------------------------------
// Router: logits -> softmax -> top-2 -> renormalized combine weights cw[T][8]
// One wave32 per token.
// ---------------------------------------------------------------------------
__global__ void router_kernel(const float* __restrict__ X,
                              const float* __restrict__ Wr,   // [H][E]
                              float* __restrict__ cw) {       // [T][E]
    int wave = threadIdx.x >> 5;
    int lane = threadIdx.x & 31;
    int t = blockIdx.x * 8 + wave;
    const float* x = X + (size_t)t * H_DIM;

    float acc[NEXP];
    #pragma unroll
    for (int e = 0; e < NEXP; ++e) acc[e] = 0.f;

    for (int h = lane; h < H_DIM; h += 32) {
        float xv = x[h];
        #pragma unroll
        for (int e = 0; e < NEXP; ++e) acc[e] += xv * Wr[h * NEXP + e];
    }
    #pragma unroll
    for (int e = 0; e < NEXP; ++e)
        #pragma unroll
        for (int off = 16; off > 0; off >>= 1)
            acc[e] += __shfl_xor(acc[e], off, 32);

    if (lane == 0) {
        float mx = acc[0];
        #pragma unroll
        for (int e = 1; e < NEXP; ++e) mx = fmaxf(mx, acc[e]);
        float p[NEXP]; float s = 0.f;
        #pragma unroll
        for (int e = 0; e < NEXP; ++e) { p[e] = __expf(acc[e] - mx); s += p[e]; }
        float inv = 1.f / s;
        #pragma unroll
        for (int e = 0; e < NEXP; ++e) p[e] *= inv;
        int i1 = 0; float w1 = p[0];
        #pragma unroll
        for (int e = 1; e < NEXP; ++e) if (p[e] > w1) { w1 = p[e]; i1 = e; }
        int i2 = (i1 == 0) ? 1 : 0; float w2 = p[i2];
        #pragma unroll
        for (int e = 0; e < NEXP; ++e)
            if (e != i1 && p[e] > w2) { w2 = p[e]; i2 = e; }
        float r = 1.f / (w1 + w2);
        #pragma unroll
        for (int e = 0; e < NEXP; ++e) {
            float v = 0.f;
            if (e == i1) v = w1 * r;
            if (e == i2) v = w2 * r;
            cw[(size_t)t * NEXP + e] = v;
        }
    }
}

// ---------------------------------------------------------------------------
// Fused MoE: for each 32-token tile, accumulate over {8 routed + shared}:
//   acc(32x1024) += [silu(X @ W1_e) * cw_e] @ W2_e, chunked over I by IC=64.
// 8 wave32 per workgroup; 128 fp32 accumulator VGPRs/thread; A fragments
// hoisted to registers, B fragments 1-deep software pipelined from L2.
// ---------------------------------------------------------------------------
__global__ void __launch_bounds__(256)
moe_kernel(const bf16_t* __restrict__ Xb,     // [T][H]
           const bf16_t* __restrict__ W1rT,   // [E][I][H]
           const bf16_t* __restrict__ W2rT,   // [E][H][I]
           const bf16_t* __restrict__ W1sT,   // [I][H]
           const bf16_t* __restrict__ W2sT,   // [H][I]
           const float*  __restrict__ cw,     // [T][E]
           float* __restrict__ out) {         // [T][H]
    extern __shared__ char smem_raw[];
    bf16_t* Xs  = (bf16_t*)smem_raw;                 // 32 x XLD
    bf16_t* Us  = Xs + M_TILE * XLD;                 // 32 x ULD
    float*  cws = (float*)(Us + M_TILE * ULD);       // 32

    const int tid  = threadIdx.x;
    const int wave = tid >> 5;
    const int lane = tid & 31;
    const int half = lane >> 4;      // K-half selector per WMMA layout
    const int l16  = lane & 15;
    const size_t trow0 = (size_t)blockIdx.x * M_TILE;

    // Stage the 32x1024 bf16 X tile into LDS (16B vector copies).
    for (int idx = tid; idx < M_TILE * (H_DIM / 8); idx += 256) {
        int r  = idx / (H_DIM / 8);
        int c8 = idx % (H_DIM / 8);
        *(v8bf*)&Xs[r * XLD + c8 * 8] =
            *(const v8bf*)&Xb[(trow0 + r) * H_DIM + c8 * 8];
    }
    __syncthreads();

    v8f acc[2][8];
    #pragma unroll
    for (int m = 0; m < 2; ++m)
        #pragma unroll
        for (int j = 0; j < 8; ++j) acc[m][j] = {};

    const int mt = wave & 1;   // U-tile row for the up phase
    const int nt = wave >> 1;  // U-tile col for the up phase

    for (int e = 0; e < NEXP + 1; ++e) {
        const bf16_t* W1 = (e < NEXP) ? W1rT + (size_t)e * I_DIM * H_DIM : W1sT;
        const bf16_t* W2 = (e < NEXP) ? W2rT + (size_t)e * H_DIM * I_DIM : W2sT;
        if (tid < M_TILE)
            cws[tid] = (e < NEXP) ? cw[(trow0 + tid) * NEXP + e] : 1.0f;
        __syncthreads();

        for (int ic = 0; ic < I_DIM; ic += IC) {
            // ---- UP: U(32xIC) = Xs(32x1024) @ W1[:, ic..ic+IC]; 1 tile/wave
            v8f u = {};
            const bf16_t* w1n =
                W1 + (size_t)(ic + nt * 16 + l16) * H_DIM + half * 16;
            const int ar = mt * 16 + l16;
            ABfrag bcur, bnxt;
            bcur.v = *(const v16bf*)&w1n[0];
            #pragma unroll 8
            for (int k = 0; k < H_DIM; k += 32) {
                ABfrag a;
                a.h[0] = *(const v8bf*)&Xs[ar * XLD + k + half * 16];
                a.h[1] = *(const v8bf*)&Xs[ar * XLD + k + half * 16 + 8];
                if (k + 32 < H_DIM)              // prefetch next B fragment
                    bnxt.v = *(const v16bf*)&w1n[k + 32];
                u = __builtin_amdgcn_wmma_f32_16x16x32_bf16(
                        false, a.v, false, bcur.v, (short)0, u, false, false);
                bcur = bnxt;
            }
            // silu * routing weight -> bf16 -> LDS (A-layout for down GEMM)
            #pragma unroll
            for (int v = 0; v < 8; ++v) {
                int row = mt * 16 + half * 8 + v;
                int col = nt * 16 + l16;
                float x = u[v];
                float s = x / (1.0f + __expf(-x));
                s *= cws[row];
                Us[row * ULD + col] = (bf16_t)s;
            }
            __syncthreads();

            // ---- DOWN: acc(32x1024) += Us(32xIC) @ W2[ic..ic+IC, :]
            // Hoist the 4 A fragments (2 row-tiles x 2 K-steps) to registers.
            ABfrag af[2][2];
            #pragma unroll
            for (int m2 = 0; m2 < 2; ++m2)
                #pragma unroll
                for (int ks2 = 0; ks2 < 2; ++ks2) {
                    int ur = m2 * 16 + l16;
                    af[m2][ks2].h[0] =
                        *(const v8bf*)&Us[ur * ULD + ks2 * 32 + half * 16];
                    af[m2][ks2].h[1] =
                        *(const v8bf*)&Us[ur * ULD + ks2 * 32 + half * 16 + 8];
                }
            // B base: row (= output col) l16 of tile, K-half offset applied.
            const bf16_t* w2base =
                W2 + ((size_t)(wave * 8 * 16 + l16)) * I_DIM + ic + half * 16;
            ABfrag bc, bn;
            bc.v = *(const v16bf*)&w2base[0];
            #pragma unroll
            for (int j = 0; j < 8; ++j) {
                #pragma unroll
                for (int ks2 = 0; ks2 < 2; ++ks2) {
                    if (!(j == 7 && ks2 == 1)) { // prefetch next B fragment
                        int nj = ks2 ? (j + 1) : j;
                        int nk = ks2 ^ 1;
                        bn.v = *(const v16bf*)&w2base[(size_t)nj * 16 * I_DIM
                                                      + nk * 32];
                    }
                    acc[0][j] = __builtin_amdgcn_wmma_f32_16x16x32_bf16(
                        false, af[0][ks2].v, false, bc.v, (short)0,
                        acc[0][j], false, false);
                    acc[1][j] = __builtin_amdgcn_wmma_f32_16x16x32_bf16(
                        false, af[1][ks2].v, false, bc.v, (short)0,
                        acc[1][j], false, false);
                    bc = bn;
                }
            }
            __syncthreads();   // Us consumed; safe to overwrite next chunk
        }
    }

    // ---- writeout: fp32, every output element written exactly once
    #pragma unroll
    for (int j = 0; j < 8; ++j) {
        const int ntile = wave * 8 + j;
        #pragma unroll
        for (int m2 = 0; m2 < 2; ++m2)
            #pragma unroll
            for (int v = 0; v < 8; ++v) {
                int row = m2 * 16 + half * 8 + v;
                int col = ntile * 16 + l16;
                out[(trow0 + row) * H_DIM + col] = acc[m2][j][v];
            }
    }
}

// ---------------------------------------------------------------------------
// Launch
// ---------------------------------------------------------------------------
extern "C" void kernel_launch(void* const* d_in, const int* in_sizes, int n_in,
                              void* d_out, int out_size, void* d_ws, size_t ws_size,
                              hipStream_t stream) {
    const float* X    = (const float*)d_in[0];   // [4,4096,H]
    const float* W1s  = (const float*)d_in[1];   // [H][I]
    const float* W2s  = (const float*)d_in[2];   // [I][H]
    const float* W1r  = (const float*)d_in[3];   // [E][H][I]
    const float* W2r  = (const float*)d_in[4];   // [E][I][H]
    const float* Wr   = (const float*)d_in[5];   // [H][E]
    float* out        = (float*)d_out;

    // Workspace carve-up (bf16 copies + combine weights), ~186 MB total.
    char* ws = (char*)d_ws;
    bf16_t* Xb   = (bf16_t*)ws;  ws += (size_t)T_TOK * H_DIM * sizeof(bf16_t);
    bf16_t* W1sT = (bf16_t*)ws;  ws += (size_t)I_DIM * H_DIM * sizeof(bf16_t);
    bf16_t* W2sT = (bf16_t*)ws;  ws += (size_t)H_DIM * I_DIM * sizeof(bf16_t);
    bf16_t* W1rT = (bf16_t*)ws;  ws += (size_t)NEXP * I_DIM * H_DIM * sizeof(bf16_t);
    bf16_t* W2rT = (bf16_t*)ws;  ws += (size_t)NEXP * H_DIM * I_DIM * sizeof(bf16_t);
    float*  cw   = (float*)ws;   ws += (size_t)T_TOK * NEXP * sizeof(float);
    (void)ws_size; (void)in_sizes; (void)n_in; (void)out_size;

    // 1) activations fp32 -> bf16
    {
        size_t n4 = (size_t)T_TOK * H_DIM / 4;
        cvt_bf16_kernel<<<(unsigned)((n4 + 255) / 256), 256, 0, stream>>>(X, Xb, n4);
    }
    // 2) weights fp32 -> bf16 with transpose (W1->[I][H], W2->[H][I])
    {
        dim3 blk(32, 8, 1);
        dim3 g1(I_DIM / 32, H_DIM / 32, 1);      // W1s: R=H, C=I -> [I][H]
        transpose_cvt_kernel<<<g1, blk, 0, stream>>>(W1s, W1sT, H_DIM, I_DIM);
        dim3 g2(H_DIM / 32, I_DIM / 32, 1);      // W2s: R=I, C=H -> [H][I]
        transpose_cvt_kernel<<<g2, blk, 0, stream>>>(W2s, W2sT, I_DIM, H_DIM);
        dim3 g3(I_DIM / 32, H_DIM / 32, NEXP);   // W1r per expert
        transpose_cvt_kernel<<<g3, blk, 0, stream>>>(W1r, W1rT, H_DIM, I_DIM);
        dim3 g4(H_DIM / 32, I_DIM / 32, NEXP);   // W2r per expert
        transpose_cvt_kernel<<<g4, blk, 0, stream>>>(W2r, W2rT, I_DIM, H_DIM);
    }
    // 3) router -> combine weights
    router_kernel<<<T_TOK / 8, 256, 0, stream>>>(X, Wr, cw);

    // 4) fused MoE GEMMs (bf16 WMMA)
    {
        size_t shmem = (size_t)(M_TILE * XLD + M_TILE * ULD) * sizeof(bf16_t)
                     + (size_t)M_TILE * sizeof(float);
        moe_kernel<<<T_TOK / M_TILE, 256, shmem, stream>>>(
            Xb, W1rT, W2rT, W1sT, W2sT, cw, out);
    }
}
